// QAConv_46488726012174
// MI455X (gfx1250) — matrix operand order repaired and checked by reading, not compiled, not run on hardware
//
#include <hip/hip_runtime.h>
#include <hip/hip_bf16.h>

// ---------------------------------------------------------------------------
// QAConv (CBAM + pairwise matching) for MI455X / gfx1250.
//
// Kernel 1: per-image CBAM (channel SE + 49x49 spatial attention) fused with
//           bf16 hi/lo split-packing into WMMA fragment order.
// Kernel 2: one block per (p,g) pair; 8 waves cover a 6x6 grid of 32x32
//           register blocks (2x2 WMMA tiles each) of the 192x192x128 GEMM
//           using v_wmma_f32_16x16x32_bf16 with a bf16x3 split for ~f32
//           accuracy. 12 WMMAs per 16 b128 loads per K-step (2x2 reuse).
//           Position-aware sliding maxes + BN + FC + sigmoid are fused into
//           the epilogue -> one float per pair; the 150 MB score tensor is
//           never materialized in HBM.
// ---------------------------------------------------------------------------

#define H_     24
#define W_     8
#define C_     128
#define HW_    192
#define N_IMG  32
#define SAK    49
#define PAD_   24
#define EPS_   1e-5f

#define TILES_PER_IMG 12           // 192 / 16
#define KBLK          4            // 128 / 32
#define FRAG_ELEMS    16           // bf16 per lane per fragment
#define IMG_PACK      (TILES_PER_IMG * KBLK * 32 * FRAG_ELEMS)  // 24576 bf16

typedef __attribute__((ext_vector_type(16))) __bf16 v16bf;
typedef __attribute__((ext_vector_type(8)))  float  v8f;

// A-matrix (16x32 bf16) per-lane K index for element e (ISA 7.12.2):
// VGPR v holds K = 16*(v>=4) + (v%4)*2 + pos  (+8 for lanes 16..31)
__device__ __forceinline__ int kmapA(int lane, int e) {
    return 16 * (e >> 3) + (((e >> 1) & 3) << 1) + (e & 1) + ((lane >> 4) << 3);
}
// B-matrix (32x16 bf16): same K order as A (K = 16*(e>=8) + (e%8) + 8*(lane>=16)).
__device__ __forceinline__ int kmapB(int lane, int e) {
    return 16 * (e >> 3) + (e & 7) + ((lane >> 4) << 3);
}

__device__ __forceinline__ float sigmoidf_(float x) {
    return 1.0f / (1.0f + __expf(-x));
}

__device__ __forceinline__ size_t frag_off(int tile, int kb, int lane) {
    return (((size_t)tile * KBLK + kb) * 32 + lane) * FRAG_ELEMS;
}

// ---------------------------------------------------------------------------
// Kernel 1: CBAM + bf16 hi/lo WMMA-fragment packing. One block per image
// (b < 32 -> probe, packed in B layout; b >= 32 -> gallery, A layout).
// ---------------------------------------------------------------------------
__global__ void __launch_bounds__(256)
cbam_pack_kernel(const float* __restrict__ prob,
                 const float* __restrict__ gal,
                 const float* __restrict__ w1,    // [8,128]
                 const float* __restrict__ w2,    // [128,8]
                 const float* __restrict__ saw,   // [2,49,49]
                 const float* __restrict__ sab,   // [1]
                 __bf16* __restrict__ packHi,
                 __bf16* __restrict__ packLo)
{
    const int b   = blockIdx.x;
    const int tid = threadIdx.x;
    const float* x = (b < N_IMG) ? (prob + (size_t)b * C_ * HW_)
                                 : (gal  + (size_t)(b - N_IMG) * C_ * HW_);

    __shared__ float chmax[C_], chavg[C_], hid[16], ca[C_];
    __shared__ float smx[HW_], sav[HW_], sa[HW_];

    // ---- channel max / avg pooling --------------------------------------
    if (tid < C_) {
        float mx = -3.4e38f, sm = 0.f;
        const float* row = x + tid * HW_;
        for (int s = 0; s < HW_; ++s) { float v = row[s]; mx = fmaxf(mx, v); sm += v; }
        chmax[tid] = mx;
        chavg[tid] = sm * (1.0f / HW_);
    }
    __syncthreads();

    // ---- SE MLP hidden (8 dims, shared weights, relu) -------------------
    if (tid < 16) {
        const int h = tid & 7;
        const float* v = (tid < 8) ? chmax : chavg;
        float acc = 0.f;
        for (int c = 0; c < C_; ++c) acc += v[c] * w1[h * C_ + c];
        hid[tid] = fmaxf(acc, 0.f);
    }
    __syncthreads();

    // ---- channel attention ----------------------------------------------
    if (tid < C_) {
        float acc = 0.f;
        #pragma unroll
        for (int h = 0; h < 8; ++h) acc += w2[tid * 8 + h] * (hid[h] + hid[8 + h]);
        ca[tid] = sigmoidf_(acc);
    }
    __syncthreads();

    // ---- spatial max / mean over channels of (x * ca) -------------------
    if (tid < HW_) {
        float mx = -3.4e38f, sm = 0.f;
        for (int c = 0; c < C_; ++c) {
            float v = x[c * HW_ + tid] * ca[c];
            mx = fmaxf(mx, v); sm += v;
        }
        smx[tid] = mx;
        sav[tid] = sm * (1.0f / C_);
    }
    __syncthreads();

    // ---- 49x49 spatial conv (pad 24 => full-image window) + sigmoid -----
    if (tid < HW_) {
        const int i = tid >> 3, j = tid & 7;
        float acc = sab[0];
        for (int r = 0; r < H_; ++r) {
            const int kh = r - i + PAD_;
            #pragma unroll
            for (int c = 0; c < W_; ++c) {
                const int kw = c - j + PAD_;
                const int s  = r * W_ + c;
                acc += smx[s] * saw[kh * SAK + kw]
                     + sav[s] * saw[SAK * SAK + kh * SAK + kw];
            }
        }
        sa[tid] = sigmoidf_(acc);
    }
    __syncthreads();

    // ---- residual CBAM output y = x*(ca*sa + 1), split into bf16 hi/lo,
    //      written directly in WMMA per-lane fragment order ----------------
    __bf16* hi = packHi + (size_t)b * IMG_PACK;
    __bf16* lo = packLo + (size_t)b * IMG_PACK;
    const bool isB = (b < N_IMG);   // probe images feed the B operand
    for (int idx = tid; idx < IMG_PACK; idx += blockDim.x) {
        const int e    = idx & 15;
        const int lane = (idx >> 4) & 31;
        const int kb   = (idx >> 9) & 3;
        const int tile = idx >> 11;
        const int kin  = isB ? kmapB(lane, e) : kmapA(lane, e);
        const int c    = kb * 32 + kin;
        const int s    = tile * 16 + (lane & 15);
        const float v  = x[c * HW_ + s] * (ca[c] * sa[s] + 1.0f);
        const __bf16 vh = (__bf16)v;
        hi[idx] = vh;
        lo[idx] = (__bf16)(v - (float)vh);
    }
}

// ---------------------------------------------------------------------------
// Per-tile fused max reduction into the LDS colmax/rmax arrays.
// C-layout (ISA 7.12.2): lane holds column N=lane&15, rows M=v+8*(lane>>4).
// Tile rows 0-7 are gal-row hr=2rt (wr=M&7), rows 8-15 are hr=2rt+1, so a
// per-lane max over the 8 acc VGPRs IS the wr-max; 8-lane shfl groups give
// the ws-max. Every (hr,s)/(r,hs) cell is owned by exactly one tile.
// ---------------------------------------------------------------------------
__device__ __forceinline__ void tile_reduce(const v8f& acc, int rt, int st, int lane,
                                            float colmax[H_][HW_],
                                            float rmax[HW_][H_])
{
    float cm = acc[0];
    #pragma unroll
    for (int v = 1; v < 8; ++v) cm = fmaxf(cm, acc[v]);
    colmax[2 * rt + (lane >> 4)][st * 16 + (lane & 15)] = cm;

    #pragma unroll
    for (int v = 0; v < 8; ++v) {
        float rv = acc[v];
        rv = fmaxf(rv, __shfl_xor(rv, 1, 32));
        rv = fmaxf(rv, __shfl_xor(rv, 2, 32));
        rv = fmaxf(rv, __shfl_xor(rv, 4, 32));
        if ((lane & 7) == v) {
            const int rrow = rt * 16 + v + ((lane >> 4) << 3);
            const int hcol = 2 * st + ((lane >> 3) & 1);
            rmax[rrow][hcol] = rv;
        }
    }
}

#define WMMA_BF16(A, B, Cacc) \
    __builtin_amdgcn_wmma_f32_16x16x32_bf16(false, (A), false, (B), (short)0, (Cacc), false, false)

// ---------------------------------------------------------------------------
// Kernel 2: one block (8 waves) per (p,g) pair. 6x6 grid of 32x32 register
// blocks; each block = 2x2 WMMA tiles, K=128 in 4 steps of 32, bf16x3 split
// -> 12 WMMAs per 16 b128 loads per K-step. __launch_bounds__(256, 1) gives
// the register allocator headroom (4 acc + 8 fragments = 96 VGPRs live) so
// nothing spills to scratch; occupancy is not the bottleneck here.
// ---------------------------------------------------------------------------
__global__ void __launch_bounds__(256, 1)
qaconv_wmma_kernel(const __bf16* __restrict__ packHi,
                   const __bf16* __restrict__ packLo,
                   const float* __restrict__ bn_g,
                   const float* __restrict__ bn_b,
                   const float* __restrict__ bn_m,
                   const float* __restrict__ bn_v,
                   const float* __restrict__ fcw,   // [192]
                   const float* __restrict__ fcb,   // [1]
                   const float* __restrict__ lbn_g,
                   const float* __restrict__ lbn_b,
                   const float* __restrict__ lbn_m,
                   const float* __restrict__ lbn_v,
                   float* __restrict__ out)
{
    const int pg   = blockIdx.x;
    const int p    = pg >> 5, g = pg & 31;
    const int tid  = threadIdx.x;
    const int wave = tid >> 5, lane = tid & 31;

    __shared__ float colmax[H_][HW_];   // [hr][s]  : max over wr      (18 KB)
    __shared__ float rmax[HW_][H_];     // [r][hs]  : max over ws      (18 KB)
    __shared__ float red[256];

    const __bf16* Ahi = packHi + (size_t)(N_IMG + g) * IMG_PACK;  // gallery -> A
    const __bf16* Alo = packLo + (size_t)(N_IMG + g) * IMG_PACK;
    const __bf16* Bhi = packHi + (size_t)p * IMG_PACK;            // probe -> B
    const __bf16* Blo = packLo + (size_t)p * IMG_PACK;

    // 36 register blocks (2x2 tiles each) over 8 waves.
    for (int t = wave; t < 36; t += 8) {
        const int rt2 = t / 6, st2 = t % 6;
        const int rt0 = 2 * rt2, rt1 = rt0 + 1;
        const int st0 = 2 * st2, st1 = st0 + 1;

        __builtin_prefetch(Ahi + frag_off(rt0, 0, lane), 0, 3);
        __builtin_prefetch(Bhi + frag_off(st0, 0, lane), 0, 3);

        v8f acc00 = {}, acc01 = {}, acc10 = {}, acc11 = {};
        #pragma unroll
        for (int kb = 0; kb < KBLK; ++kb) {
            const size_t a0 = frag_off(rt0, kb, lane);
            const size_t a1 = frag_off(rt1, kb, lane);
            const size_t b0 = frag_off(st0, kb, lane);
            const size_t b1 = frag_off(st1, kb, lane);
            const v16bf a_hi0 = *(const v16bf*)(Ahi + a0);
            const v16bf a_lo0 = *(const v16bf*)(Alo + a0);
            const v16bf a_hi1 = *(const v16bf*)(Ahi + a1);
            const v16bf a_lo1 = *(const v16bf*)(Alo + a1);
            const v16bf b_hi0 = *(const v16bf*)(Bhi + b0);
            const v16bf b_lo0 = *(const v16bf*)(Blo + b0);
            const v16bf b_hi1 = *(const v16bf*)(Bhi + b1);
            const v16bf b_lo1 = *(const v16bf*)(Blo + b1);

            // bf16x3 split precision: hi*hi + hi*lo + lo*hi, f32 accumulate.
            acc00 = WMMA_BF16(a_hi0, b_hi0, acc00);
            acc00 = WMMA_BF16(a_hi0, b_lo0, acc00);
            acc00 = WMMA_BF16(a_lo0, b_hi0, acc00);

            acc01 = WMMA_BF16(a_hi0, b_hi1, acc01);
            acc01 = WMMA_BF16(a_hi0, b_lo1, acc01);
            acc01 = WMMA_BF16(a_lo0, b_hi1, acc01);

            acc10 = WMMA_BF16(a_hi1, b_hi0, acc10);
            acc10 = WMMA_BF16(a_hi1, b_lo0, acc10);
            acc10 = WMMA_BF16(a_lo1, b_hi0, acc10);

            acc11 = WMMA_BF16(a_hi1, b_hi1, acc11);
            acc11 = WMMA_BF16(a_hi1, b_lo1, acc11);
            acc11 = WMMA_BF16(a_lo1, b_hi1, acc11);
        }

        tile_reduce(acc00, rt0, st0, lane, colmax, rmax);
        tile_reduce(acc01, rt0, st1, lane, colmax, rmax);
        tile_reduce(acc10, rt1, st0, lane, colmax, rmax);
        tile_reduce(acc11, rt1, st1, lane, colmax, rmax);
    }
    __syncthreads();

    // ---- windowed max (part_rows=6, start=clip(h-3,0,18)) + BN + FC -----
    const float bnscale = bn_g[0] * rsqrtf(bn_v[0] + EPS_);
    const float bnshift = bn_b[0] - bn_m[0] * bnscale;

    float partial = 0.f;
    if (tid < HW_) {
        const int hs = tid >> 3;
        int a0 = hs - 3; a0 = a0 < 0 ? 0 : (a0 > 18 ? 18 : a0);
        float m1 = -3.4e38f;
        #pragma unroll
        for (int hr = 0; hr < 6; ++hr) m1 = fmaxf(m1, colmax[a0 + hr][tid]);

        const int hr = tid >> 3;
        int a1 = hr - 3; a1 = a1 < 0 ? 0 : (a1 > 18 ? 18 : a1);
        float m2 = -3.4e38f;
        #pragma unroll
        for (int h2 = 0; h2 < 6; ++h2) m2 = fmaxf(m2, rmax[tid][a1 + h2]);

        partial = ((m1 * bnscale + bnshift) + (m2 * bnscale + bnshift)) * fcw[tid];
    }
    red[tid] = partial;
    __syncthreads();
    #pragma unroll
    for (int s = 128; s > 0; s >>= 1) {
        if (tid < s) red[tid] += red[tid + s];
        __syncthreads();
    }
    if (tid == 0) {
        float y = red[0] + 2.0f * fcb[0];
        const float ls = lbn_g[0] * rsqrtf(lbn_v[0] + EPS_);
        y = (y - lbn_m[0]) * ls + lbn_b[0];
        out[pg] = sigmoidf_(y);
    }
}

// ---------------------------------------------------------------------------
extern "C" void kernel_launch(void* const* d_in, const int* in_sizes, int n_in,
                              void* d_out, int out_size, void* d_ws, size_t ws_size,
                              hipStream_t stream)
{
    const float* prob  = (const float*)d_in[0];
    const float* gal   = (const float*)d_in[1];
    const float* se_w1 = (const float*)d_in[2];
    const float* se_w2 = (const float*)d_in[3];
    const float* sa_w  = (const float*)d_in[4];
    const float* sa_b  = (const float*)d_in[5];
    const float* bn_g  = (const float*)d_in[6];
    const float* bn_b  = (const float*)d_in[7];
    const float* bn_m  = (const float*)d_in[8];
    const float* bn_v  = (const float*)d_in[9];
    const float* fc_w  = (const float*)d_in[10];
    const float* fc_b  = (const float*)d_in[11];
    const float* lbn_g = (const float*)d_in[12];
    const float* lbn_b = (const float*)d_in[13];
    const float* lbn_m = (const float*)d_in[14];
    const float* lbn_v = (const float*)d_in[15];

    // Workspace: 64 images x 24576 bf16, hi then lo (~6.3 MB total).
    __bf16* packHi = (__bf16*)d_ws;
    __bf16* packLo = (__bf16*)((char*)d_ws + (size_t)64 * IMG_PACK * sizeof(__bf16));

    cbam_pack_kernel<<<64, 256, 0, stream>>>(prob, gal, se_w1, se_w2, sa_w, sa_b,
                                             packHi, packLo);
    qaconv_wmma_kernel<<<32 * 32, 256, 0, stream>>>(packHi, packLo,
                                                    bn_g, bn_b, bn_m, bn_v,
                                                    fc_w, fc_b,
                                                    lbn_g, lbn_b, lbn_m, lbn_v,
                                                    (float*)d_out);
}
